// FusionLayer_36378372997497
// MI455X (gfx1250) — matrix operand order repaired and checked by reading, not compiled, not run on hardware
//
#include <hip/hip_runtime.h>
#include <hip/hip_bf16.h>

#define EPSF 1e-5f

typedef __attribute__((ext_vector_type(16))) _Float16 v16h;
typedef __attribute__((ext_vector_type(8)))  float    v8f;
typedef __attribute__((ext_vector_type(8)))  _Float16 h8;

static __device__ __forceinline__ float sigmoidf_(float x){ return 1.f/(1.f+__expf(-x)); }
static __device__ __forceinline__ float siluf_(float x){ return x*sigmoidf_(x); }
static __device__ __forceinline__ float softplusf_(float x){ return (x>30.f)?x:log1pf(__expf(x)); }

// ---------------- concat (C,L) channel-major inputs -> (L,32) position-major ----------
__global__ void k_concat(const float* __restrict__ l, const float* __restrict__ s,
                         float* __restrict__ x0, int L){
  int idx = blockIdx.x*blockDim.x + threadIdx.x;
  if (idx >= L*32) return;
  int pos = idx >> 5; int c = idx & 31;
  float v = (c < 16) ? l[(size_t)c*L + pos] : s[(size_t)(c-16)*L + pos];
  x0[(size_t)pos*32 + c] = v;
}

// ---------------- f32 -> f16 weight convert with column zero-pad ----------------------
__global__ void k_wcvt(const float* __restrict__ src, _Float16* __restrict__ dst,
                       int R, int C, int CP){
  int idx = blockIdx.x*blockDim.x + threadIdx.x;
  if (idx >= R*CP) return;
  int r = idx / CP, c = idx - r*CP;
  dst[idx] = (c < C) ? (_Float16)src[(size_t)r*C + c] : (_Float16)0.f;
}

// ---------------- LayerNorm over C channels, emit f16 padded to CP --------------------
template<int C, int CP>
__global__ void k_ln(const float* __restrict__ x, const float* __restrict__ w,
                     const float* __restrict__ b, _Float16* __restrict__ out, int L){
  int l = blockIdx.x*blockDim.x + threadIdx.x;
  if (l >= L) return;
  const float* row = x + (size_t)l*C;
  float v[C];
  float mu = 0.f;
  #pragma unroll
  for (int c = 0; c < C; ++c){ v[c] = row[c]; mu += v[c]; }
  mu *= (1.f/C);
  float var = 0.f;
  #pragma unroll
  for (int c = 0; c < C; ++c){ float d = v[c]-mu; var += d*d; }
  var *= (1.f/C);
  float r = rsqrtf(var + EPSF);
  _Float16* o = out + (size_t)l*CP;
  #pragma unroll
  for (int c = 0; c < CP; ++c)
    o[c] = (c < C) ? (_Float16)((v[c]-mu)*r*w[c] + b[c]) : (_Float16)0.f;
}

// ---------------- WMMA GEMM: C(MxN,f32) = A(MxK,f16) * Bw(NxK,f16)^T [+ res] ----------
// One wave per 16x16 output tile; K compile-time multiple of 32; ISA lane layouts:
//   A: m=lane&15, k = hi*8 + j (j<8) / 16 + hi*8 + (j-8)
//   B: n=lane&15, k = hi*8 + j (j<8) / 16 + hi*8 + (j-8)  (same as A)
//   C: n=lane&15, row = 8*hi + r
// RES templated so the epilogue is branch-free (no per-element null checks).
template<int K, bool RES>
__global__ void k_gemm_wmma(const _Float16* __restrict__ A, const _Float16* __restrict__ Bw,
                            float* __restrict__ C, const float* __restrict__ res,
                            int M, int N){
  int wave = (blockIdx.x*blockDim.x + threadIdx.x) >> 5;
  int lane = threadIdx.x & 31;
  int ntiles = N >> 4;
  int mt = wave / ntiles;
  int nt = wave - mt*ntiles;
  if (mt*16 >= M) return;              // wave-uniform: EXEC stays all-1s for WMMA
  int m = lane & 15, hi = lane >> 4;
  v8f acc = {0.f,0.f,0.f,0.f,0.f,0.f,0.f,0.f};
  #pragma unroll
  for (int k0 = 0; k0 < K; k0 += 32){
    const _Float16* ap = A  + (size_t)(mt*16 + m)*K + k0 + hi*8;
    const _Float16* bp = Bw + (size_t)(nt*16 + m)*K + k0 + hi*8;
    h8 alo = *(const h8*)(ap);
    h8 ahi = *(const h8*)(ap + 16);
    h8 blo = *(const h8*)(bp);
    h8 bhi = *(const h8*)(bp + 16);
    v16h a, b;
    #pragma unroll
    for (int j = 0; j < 8; ++j){ a[j]=alo[j]; a[j+8]=ahi[j]; b[j]=blo[j]; b[j+8]=bhi[j]; }
    acc = __builtin_amdgcn_wmma_f32_16x16x32_f16(false, a, false, b, (short)0, acc, false, false);
  }
  size_t base = (size_t)(mt*16 + 8*hi)*N + (nt*16 + m);
  if (RES){
    float r8[8];
    #pragma unroll
    for (int r = 0; r < 8; ++r) r8[r] = res[base + (size_t)r*N];
    #pragma unroll
    for (int r = 0; r < 8; ++r) C[base + (size_t)r*N] = acc[r] + r8[r];
  } else {
    #pragma unroll
    for (int r = 0; r < 8; ++r) C[base + (size_t)r*N] = acc[r];
  }
}

// ---------------- depthwise causal conv (k=4, left pad 3) + SiLU ----------------------
__global__ void k_dwconv_silu(const float* __restrict__ xz, const float* __restrict__ cw,
                              const float* __restrict__ cb, float* __restrict__ xc,
                              int L, int di, int ldxz){
  int idx = blockIdx.x*blockDim.x + threadIdx.x;
  if (idx >= L*di) return;
  int l = idx / di; int d = idx - l*di;
  float acc = cb[d];
  #pragma unroll
  for (int j = 0; j < 4; ++j){
    int t = l - 3 + j;
    if (t >= 0) acc += cw[d*4 + j] * xz[(size_t)t*ldxz + d];
  }
  xc[(size_t)l*di + d] = siluf_(acc);
}

// ---------------- x-projection: dbc = xp_w @ xc ; dt = softplus(dt_w@dbc[:dtr]+dt_b) ---
template<int DI, int DTR>
__global__ void k_xproj(const float* __restrict__ xc, const float* __restrict__ xpw,
                        const float* __restrict__ dtw, const float* __restrict__ dtb,
                        float* __restrict__ dtv, float* __restrict__ Bo,
                        float* __restrict__ Co, int L){
  int l = blockIdx.x*blockDim.x + threadIdx.x;
  if (l >= L) return;
  float v[DI];
  #pragma unroll
  for (int i = 0; i < DI; ++i) v[i] = xc[(size_t)l*DI + i];
  float dbc[DTR + 32];
  #pragma unroll
  for (int o = 0; o < DTR + 32; ++o){
    const float* wr = xpw + (size_t)o*DI;
    float acc = 0.f;
    #pragma unroll
    for (int i = 0; i < DI; ++i) acc += wr[i]*v[i];
    dbc[o] = acc;
  }
  #pragma unroll
  for (int d = 0; d < DI; ++d){
    float acc = dtb[d];
    #pragma unroll
    for (int r = 0; r < DTR; ++r) acc += dtw[d*DTR + r]*dbc[r];
    dtv[(size_t)l*DI + d] = softplusf_(acc);
  }
  #pragma unroll
  for (int s = 0; s < 16; ++s){
    Bo[(size_t)l*16 + s] = dbc[DTR + s];
    Co[(size_t)l*16 + s] = dbc[DTR + 16 + s];
  }
}

// ---------------- chunked selective scan -----------------------------------------------
// Recurrence h[t] = a[t]*h[t-1] + b[t],  a = exp(dt*A), b = dt*xc*B.
// Pass A: per segment, per (d,s) lane: P = prod(a), Hl = local scan end (h0=0).
// Pass B: tiny serial scan over segments per (d,s): Hin[j+1] = P[j]*Hin[j] + Hl[j]
//         (stored in-place as the exclusive incoming state).
// Pass C: re-run each segment from Hin, contract over s (shfl tree), emit y.
__global__ void k_scan_part(const float* __restrict__ dtv, const float* __restrict__ xc,
                            const float* __restrict__ Bo,
                            const float* __restrict__ A_log,
                            float* __restrict__ Pseg, float* __restrict__ Hseg,
                            int L, int di, int T, int S){
  int wave = (blockIdx.x*blockDim.x + threadIdx.x) >> 5;
  int lane = threadIdx.x & 31;
  int half = lane >> 4, s = lane & 15;
  int wpseg = di >> 1;                    // waves per segment
  int seg = wave / wpseg;
  int wq  = wave - seg*wpseg;
  if (seg >= S) return;                   // wave-uniform
  int d = wq*2 + half;
  float A = -__expf(A_log[(size_t)d*16 + s]);
  float P = 1.f, h = 0.f;
  int t0 = seg*T, t1 = t0 + T;
  for (int t = t0; t < t1; ++t){
    float dt = dtv[(size_t)t*di + d];
    float xv = xc[(size_t)t*di + d];
    float b  = Bo[(size_t)t*16 + s];
    float a  = __expf(dt*A);
    P *= a;
    h = a*h + (dt*xv)*b;
  }
  size_t o = (size_t)seg*di*16 + (size_t)d*16 + s;
  Pseg[o] = P;
  Hseg[o] = h;
}

__global__ void k_scan_combine(const float* __restrict__ Pseg, float* __restrict__ Hseg,
                               int di, int S){
  int idx = blockIdx.x*blockDim.x + threadIdx.x;   // (d,s) pair
  if (idx >= di*16) return;
  float hin = 0.f;
  for (int j = 0; j < S; ++j){
    size_t o = (size_t)j*di*16 + idx;
    float nxt = Pseg[o]*hin + Hseg[o];
    Hseg[o] = hin;                                  // exclusive incoming state
    hin = nxt;
  }
}

__global__ void k_scan_final(const float* __restrict__ dtv, const float* __restrict__ xc,
                             const float* __restrict__ Bo, const float* __restrict__ Co,
                             const float* __restrict__ A_log, const float* __restrict__ Dp,
                             const float* __restrict__ Hseg, float* __restrict__ y,
                             int L, int di, int T, int S){
  int wave = (blockIdx.x*blockDim.x + threadIdx.x) >> 5;
  int lane = threadIdx.x & 31;
  int half = lane >> 4, s = lane & 15;
  int wpseg = di >> 1;
  int seg = wave / wpseg;
  int wq  = wave - seg*wpseg;
  if (seg >= S) return;                   // wave-uniform
  int d = wq*2 + half;
  float A  = -__expf(A_log[(size_t)d*16 + s]);
  float Dd = Dp[d];
  float h  = Hseg[(size_t)seg*di*16 + (size_t)d*16 + s];
  int t0 = seg*T, t1 = t0 + T;
  for (int t = t0; t < t1; ++t){
    float dt = dtv[(size_t)t*di + d];
    float xv = xc[(size_t)t*di + d];
    float b  = Bo[(size_t)t*16 + s];
    float c  = Co[(size_t)t*16 + s];
    h = __expf(dt*A)*h + (dt*xv)*b;
    float contrib = h*c;
    contrib += __shfl_xor(contrib, 8, 16);
    contrib += __shfl_xor(contrib, 4, 16);
    contrib += __shfl_xor(contrib, 2, 16);
    contrib += __shfl_xor(contrib, 1, 16);
    if (s == 0) y[(size_t)t*di + d] = contrib + xv*Dd;
    if (t + 32 < t1) __builtin_prefetch(&Bo[(size_t)(t+32)*16 + s], 0, 1);
  }
}

// ---------------- gate: g = f16( y * silu(z) ) ----------------------------------------
__global__ void k_gate(const float* __restrict__ y, const float* __restrict__ xz,
                       _Float16* __restrict__ g, int L, int di, int ldxz){
  int idx = blockIdx.x*blockDim.x + threadIdx.x;
  if (idx >= L*di) return;
  int l = idx / di, d = idx - l*di;
  float z = xz[(size_t)l*ldxz + di + d];
  g[idx] = (_Float16)(y[idx] * siluf_(z));
}

// ---------------- 3x3x3 conv, pad 1, input/output position-major (L,C) ----------------
template<int CIN>
__global__ void k_conv3d(const float* __restrict__ in, const float* __restrict__ w,
                         const float* __restrict__ bias, float* __restrict__ out,
                         int Cout, int Dd, int Hh, int Ww){
  int idx = blockIdx.x*blockDim.x + threadIdx.x;
  int Lh = Dd*Hh*Ww;
  if (idx >= Lh*Cout) return;
  int oc = idx % Cout; int pos = idx / Cout;
  int xq = pos % Ww; int t = pos / Ww; int yq = t % Hh; int zq = t / Hh;
  float acc = bias[oc];
  for (int dz = -1; dz <= 1; ++dz){ int zz = zq+dz; if ((unsigned)zz >= (unsigned)Dd) continue;
    for (int dy = -1; dy <= 1; ++dy){ int yy = yq+dy; if ((unsigned)yy >= (unsigned)Hh) continue;
      for (int dx = -1; dx <= 1; ++dx){ int xx = xq+dx; if ((unsigned)xx >= (unsigned)Ww) continue;
        int p2 = (zz*Hh + yy)*Ww + xx;
        const float* ip = in + (size_t)p2*CIN;
        int koff = ((dz+1)*3 + (dy+1))*3 + (dx+1);
        const float* wp = w + ((size_t)oc*CIN)*27 + koff;
        #pragma unroll
        for (int ic = 0; ic < CIN; ++ic) acc += wp[(size_t)ic*27] * ip[ic];
      } } }
  out[(size_t)pos*Cout + oc] = acc;
}

// ---------------- batch-norm: per-channel mean/var over L, then apply -----------------
__global__ void k_bnstats(const float* __restrict__ in, float* __restrict__ stats,
                          int L, int C){
  int c = blockIdx.x;
  float s = 0.f, s2 = 0.f;
  for (int l = threadIdx.x; l < L; l += blockDim.x){
    float v = in[(size_t)l*C + c];
    s += v; s2 += v*v;
  }
  __shared__ float sh[256], sh2[256];
  sh[threadIdx.x] = s; sh2[threadIdx.x] = s2;
  __syncthreads();
  for (int o = 128; o > 0; o >>= 1){
    if ((int)threadIdx.x < o){ sh[threadIdx.x] += sh[threadIdx.x+o]; sh2[threadIdx.x] += sh2[threadIdx.x+o]; }
    __syncthreads();
  }
  if (threadIdx.x == 0){
    float mean = sh[0]/L;
    stats[c*2]   = mean;
    stats[c*2+1] = sh2[0]/L - mean*mean;
  }
}

__global__ void k_bnapply(const float* __restrict__ in, const float* __restrict__ stats,
                          const float* __restrict__ g, const float* __restrict__ b,
                          float* __restrict__ out, int L, int C, int chanMajorOut){
  int idx = blockIdx.x*blockDim.x + threadIdx.x;
  if (idx >= L*C) return;
  int l = idx / C, c = idx - l*C;
  float mean = stats[c*2], var = stats[c*2+1];
  float v = (in[idx]-mean)*rsqrtf(var + EPSF)*g[c] + b[c];
  if (chanMajorOut) out[(size_t)c*L + l] = v;
  else              out[idx] = v;
}

extern "C" void kernel_launch(void* const* d_in, const int* in_sizes, int n_in,
                              void* d_out, int out_size, void* d_ws, size_t ws_size,
                              hipStream_t stream){
  (void)in_sizes; (void)n_in; (void)out_size; (void)ws_size;
  const int L = 16*48*48;            // 36864 sequence positions
  const int S = 144;                 // scan segments
  const int T = L / S;               // 256 steps per segment
  const int TB = 256;
  auto cdiv = [](long a, long b){ return (int)((a + b - 1)/b); };

  const float* lp = (const float*)d_in[0];
  const float* sp = (const float*)d_in[1];
  // m1: 2..12  m2: 13..23  conv/bn: 24..31 (setup_inputs dict order)

  // ---- workspace arena (~77 MB; fits the 192 MB L2 -> HBM never in the loop) ----
  char* ws = (char*)d_ws; size_t off = 0;
  auto alloc = [&](size_t bytes)->char*{ char* p = ws + off; off += (bytes + 255) & ~(size_t)255; return p; };
  float*    x0    = (float*)alloc((size_t)L*32*4);   // concat input / m1 residual
  float*    x1    = (float*)alloc((size_t)L*32*4);   // mamba1 out + residual
  float*    x2    = (float*)alloc((size_t)L*16*4);   // bn1 out / m2 residual
  float*    x3    = (float*)alloc((size_t)L*16*4);   // mamba2 out + residual
  float*    cv    = (float*)alloc((size_t)L*16*4);   // raw conv outputs
  float*    stats = (float*)alloc(256);
  _Float16* w_in1  = (_Float16*)alloc(128*32*2);
  _Float16* w_out1 = (_Float16*)alloc(32*64*2);
  _Float16* w_in2  = (_Float16*)alloc(64*32*2);      // (64,16) zero-padded to K=32
  _Float16* w_out2 = (_Float16*)alloc(16*32*2);
  _Float16* s_xn = (_Float16*)alloc((size_t)L*32*2); // shared between both mamba blocks
  float*    s_xz = (float*)alloc((size_t)L*128*4);
  float*    s_xc = (float*)alloc((size_t)L*64*4);
  float*    s_dt = (float*)alloc((size_t)L*64*4);
  float*    s_B  = (float*)alloc((size_t)L*16*4);
  float*    s_C  = (float*)alloc((size_t)L*16*4);
  float*    s_y  = (float*)alloc((size_t)L*64*4);
  _Float16* s_g  = (_Float16*)alloc((size_t)L*64*2);
  float*    s_P  = (float*)alloc((size_t)S*64*16*4); // segment products
  float*    s_H  = (float*)alloc((size_t)S*64*16*4); // segment states (-> incoming)

  // ---- stage 0: concat + weight conversion ----
  k_concat<<<cdiv((long)L*32,TB),TB,0,stream>>>(lp, sp, x0, L);
  k_wcvt<<<cdiv(128*32,TB),TB,0,stream>>>((const float*)d_in[4],  w_in1, 128, 32, 32);
  k_wcvt<<<cdiv(32*64,TB),TB,0,stream>>>((const float*)d_in[12], w_out1, 32, 64, 64);
  k_wcvt<<<cdiv(64*32,TB),TB,0,stream>>>((const float*)d_in[15], w_in2,  64, 16, 32);
  k_wcvt<<<cdiv(16*32,TB),TB,0,stream>>>((const float*)d_in[23], w_out2, 16, 32, 32);

  // ---- mamba block 1 (dim 32, di 64, dtr 2) ----
  k_ln<32,32><<<cdiv(L,TB),TB,0,stream>>>(x0, (const float*)d_in[2], (const float*)d_in[3], s_xn, L);
  { long waves = (long)(L/16)*(128/16);
    k_gemm_wmma<32,false><<<cdiv(waves*32,TB),TB,0,stream>>>(s_xn, w_in1, s_xz, nullptr, L, 128); }
  k_dwconv_silu<<<cdiv((long)L*64,TB),TB,0,stream>>>(s_xz, (const float*)d_in[5], (const float*)d_in[6], s_xc, L, 64, 128);
  k_xproj<64,2><<<cdiv(L,TB),TB,0,stream>>>(s_xc, (const float*)d_in[7], (const float*)d_in[8], (const float*)d_in[9], s_dt, s_B, s_C, L);
  { long waves = (long)S*(64/2);
    k_scan_part<<<cdiv(waves*32,TB),TB,0,stream>>>(s_dt, s_xc, s_B, (const float*)d_in[10], s_P, s_H, L, 64, T, S);
    k_scan_combine<<<cdiv(64*16,TB),TB,0,stream>>>(s_P, s_H, 64, S);
    k_scan_final<<<cdiv(waves*32,TB),TB,0,stream>>>(s_dt, s_xc, s_B, s_C, (const float*)d_in[10], (const float*)d_in[11], s_H, s_y, L, 64, T, S); }
  k_gate<<<cdiv((long)L*64,TB),TB,0,stream>>>(s_y, s_xz, s_g, L, 64, 128);
  { long waves = (long)(L/16)*(32/16);
    k_gemm_wmma<64,true><<<cdiv(waves*32,TB),TB,0,stream>>>(s_g, w_out1, x1, x0, L, 32); }

  // ---- conv1 + bn1 ----
  k_conv3d<32><<<cdiv((long)L*16,TB),TB,0,stream>>>(x1, (const float*)d_in[24], (const float*)d_in[25], cv, 16, 16, 48, 48);
  k_bnstats<<<16,TB,0,stream>>>(cv, stats, L, 16);
  k_bnapply<<<cdiv((long)L*16,TB),TB,0,stream>>>(cv, stats, (const float*)d_in[26], (const float*)d_in[27], x2, L, 16, 0);

  // ---- mamba block 2 (dim 16, di 32, dtr 1; K padded to 32) ----
  k_ln<16,32><<<cdiv(L,TB),TB,0,stream>>>(x2, (const float*)d_in[13], (const float*)d_in[14], s_xn, L);
  { long waves = (long)(L/16)*(64/16);
    k_gemm_wmma<32,false><<<cdiv(waves*32,TB),TB,0,stream>>>(s_xn, w_in2, s_xz, nullptr, L, 64); }
  k_dwconv_silu<<<cdiv((long)L*32,TB),TB,0,stream>>>(s_xz, (const float*)d_in[16], (const float*)d_in[17], s_xc, L, 32, 64);
  k_xproj<32,1><<<cdiv(L,TB),TB,0,stream>>>(s_xc, (const float*)d_in[18], (const float*)d_in[19], (const float*)d_in[20], s_dt, s_B, s_C, L);
  { long waves = (long)S*(32/2);
    k_scan_part<<<cdiv(waves*32,TB),TB,0,stream>>>(s_dt, s_xc, s_B, (const float*)d_in[21], s_P, s_H, L, 32, T, S);
    k_scan_combine<<<cdiv(32*16,TB),TB,0,stream>>>(s_P, s_H, 32, S);
    k_scan_final<<<cdiv(waves*32,TB),TB,0,stream>>>(s_dt, s_xc, s_B, s_C, (const float*)d_in[21], (const float*)d_in[22], s_H, s_y, L, 32, T, S); }
  k_gate<<<cdiv((long)L*32,TB),TB,0,stream>>>(s_y, s_xz, s_g, L, 32, 64);
  { long waves = (long)(L/16)*(16/16);
    k_gemm_wmma<32,true><<<cdiv(waves*32,TB),TB,0,stream>>>(s_g, w_out2, x3, x2, L, 16); }

  // ---- conv2 + bn2 -> d_out (NCDHW channel-major) ----
  k_conv3d<16><<<cdiv((long)L*16,TB),TB,0,stream>>>(x3, (const float*)d_in[28], (const float*)d_in[29], cv, 16, 16, 48, 48);
  k_bnstats<<<16,TB,0,stream>>>(cv, stats, L, 16);
  k_bnapply<<<cdiv((long)L*16,TB),TB,0,stream>>>(cv, stats, (const float*)d_in[30], (const float*)d_in[31], (float*)d_out, L, 16, 1);
}
